// LlamaTrunkBeforeNorm_20160576488108
// MI455X (gfx1250) — compile-verified, matrix operand order checked by reading, not run on hardware
//
#include <hip/hip_runtime.h>

typedef __attribute__((ext_vector_type(16))) __bf16 v16bf;
typedef __attribute__((ext_vector_type(8)))  __bf16 v8bf;
typedef __attribute__((ext_vector_type(8)))  float  v8f;

#define DEV __device__ __forceinline__

static constexpr int LAYERS = 2, NH = 32, NKV = 8, HD = 64;
static constexpr int DMODEL = 2048, FFD = 8192, SEQ = 1024;
static constexpr float EPSV = 1e-5f;

union F16U { v16bf v; v8bf h[2]; };

// Load one 16x32 bf16 WMMA fragment.  The identical lane pattern serves as
// the A fragment from a row-major [m][k] source and as the B fragment from an
// N-major [n][k] source.  Two 16-byte loads per lane.
DEV v16bf load_frag16(const __bf16* base, int ld) {
  const int lane = threadIdx.x & 31;
  const int r = lane & 15;
  const int hi = (lane >> 4) << 3;          // 0 or 8
  F16U f;
  f.h[0] = *(const v8bf*)(base + (size_t)r * ld + hi);
  f.h[1] = *(const v8bf*)(base + (size_t)r * ld + 16 + hi);
  return f.v;
}

DEV v8f wmma_bf16(v16bf a, v16bf b, v8f c) {
  return __builtin_amdgcn_wmma_f32_16x16x32_bf16(false, a, false, b, (short)0, c, false, false);
}

// ---------------------------------------------------------------- embedding
__global__ __launch_bounds__(256)
void embed_gather(const int* __restrict__ ids, const float* __restrict__ emb,
                  float* __restrict__ x) {
  const int s = blockIdx.x;
  const float4* src = (const float4*)(emb + (size_t)ids[s] * DMODEL);
  float4* dst = (float4*)(x + (size_t)s * DMODEL);
  for (int i = threadIdx.x; i < DMODEL / 4; i += 256) dst[i] = src[i];
}

// ------------------------------------------------- weight transpose+convert
// src[K][N] fp32 (row-major)  ->  dst[N][K] bf16.  Coalesced both directions
// through a padded LDS tile.  Runs once per layer (~HBM stream rate).
__global__ __launch_bounds__(256)
void transpose_bf16(const float* __restrict__ src, __bf16* __restrict__ dst,
                    int K, int N) {
  __shared__ float t[32][33];
  const int nb = blockIdx.x * 32, kb = blockIdx.y * 32;
  const int tx = threadIdx.x & 31, ty = threadIdx.x >> 5;   // ty: 0..7
  #pragma unroll
  for (int i = 0; i < 4; ++i)
    t[ty + 8 * i][tx] = src[(size_t)(kb + ty + 8 * i) * N + nb + tx];
  __syncthreads();
  #pragma unroll
  for (int i = 0; i < 4; ++i)
    dst[(size_t)(nb + ty + 8 * i) * K + kb + tx] = (__bf16)t[tx][ty + 8 * i];
}

// ---------------------------------------------------------------- rmsnorm
__global__ __launch_bounds__(256)
void rmsnorm_bf16(const float* __restrict__ x, const float* __restrict__ g,
                  __bf16* __restrict__ h) {
  __shared__ float red[256];
  const int row = blockIdx.x, tid = threadIdx.x;
  const float* xr = x + (size_t)row * DMODEL;
  float ss = 0.f;
  for (int i = tid; i < DMODEL; i += 256) { float v = xr[i]; ss += v * v; }
  red[tid] = ss; __syncthreads();
  for (int s = 128; s > 0; s >>= 1) {
    if (tid < s) red[tid] += red[tid + s];
    __syncthreads();
  }
  const float sc = rsqrtf(red[0] * (1.f / DMODEL) + EPSV);
  __bf16* hr = h + (size_t)row * DMODEL;
  for (int i = tid; i < DMODEL; i += 256) hr[i] = (__bf16)(xr[i] * sc * g[i]);
}

// ---------------------------------------------------------------- GEMM (TN)
// O[M,N] = A_bf16[M,K] @ Bt_bf16[N,K]^T  (+ residual).
// Block 128x128, 8 waves; wave = 32Mx64N (2 A-frags x 4 B-frags, 8 WMMA per
// 32-K step).  No LDS: both operand fragments stream from global/L2 as b128.
template <bool RES>
__global__ __launch_bounds__(256)
void gemm_tn(const __bf16* __restrict__ A, const __bf16* __restrict__ Bt,
             float* __restrict__ O, const float* __restrict__ Rsd,
             int M, int N, int K) {
  const int wave = threadIdx.x >> 5, lane = threadIdx.x & 31;
  const int m0 = blockIdx.y * 128 + (wave >> 1) * 32;
  const int n0 = blockIdx.x * 128 + (wave & 1) * 64;
  v8f acc[2][4] = {{{}, {}, {}, {}}, {{}, {}, {}, {}}};
  const __bf16* Ap = A + (size_t)m0 * K;
  const __bf16* Bp = Bt + (size_t)n0 * K;
  #pragma unroll 2
  for (int kk = 0; kk < K; kk += 32) {
    const v16bf a0 = load_frag16(Ap + kk, K);
    const v16bf a1 = load_frag16(Ap + (size_t)16 * K + kk, K);
    #pragma unroll
    for (int t = 0; t < 4; ++t) {
      const v16bf b = load_frag16(Bp + (size_t)(t * 16) * K + kk, K);
      acc[0][t] = wmma_bf16(a0, b, acc[0][t]);
      acc[1][t] = wmma_bf16(a1, b, acc[1][t]);
    }
  }
  const int c = lane & 15, hi8 = (lane >> 4) * 8;
  #pragma unroll
  for (int u = 0; u < 2; ++u)
    #pragma unroll
    for (int t = 0; t < 4; ++t)
      #pragma unroll
      for (int j = 0; j < 8; ++j) {
        const size_t idx = (size_t)(m0 + u * 16 + hi8 + j) * N + n0 + t * 16 + c;
        float v = acc[u][t][j];
        if (RES) v += Rsd[idx];
        O[idx] = v;
      }
}

// ---------------------------------------------------------------- RoPE+pack
// qkv: fused GEMM output [S][3072] fp32 (q | k | v).  Produce WMMA layouts:
//   qh: [H][S][64]   kh: [KVH][S][64]  (rope applied, HF half-rotate)
//   vt: [KVH][64][S] (transposed so PV B-fragments are contiguous)
__global__ __launch_bounds__(64)
void rope_pack(const float* __restrict__ qkv, __bf16* __restrict__ qh,
               __bf16* __restrict__ kh, __bf16* __restrict__ vt) {
  const int s = blockIdx.x, slot = blockIdx.y, d = threadIdx.x;
  const float ang = s * __powf(10000.f, -(2.f * (d & 31)) / 64.f);
  const float cs = __cosf(ang), sn = __sinf(ang);
  const float* row = qkv + (size_t)s * 3072;
  if (slot < NH) {
    const float* src = row + slot * HD;
    const float t = src[d];
    const float rot = (d < 32) ? -src[d + 32] : src[d - 32];
    qh[((size_t)slot * SEQ + s) * HD + d] = (__bf16)(t * cs + rot * sn);
  } else if (slot < NH + NKV) {
    const int hh = slot - NH;
    const float* src = row + 2048 + hh * HD;
    const float t = src[d];
    const float rot = (d < 32) ? -src[d + 32] : src[d - 32];
    kh[((size_t)hh * SEQ + s) * HD + d] = (__bf16)(t * cs + rot * sn);
  } else {
    const int hh = slot - NH - NKV;
    vt[((size_t)hh * HD + d) * SEQ + s] = (__bf16)row[2560 + hh * HD + d];
  }
}

// ---------------------------------------------------------------- flash attn
// grid (H, S/128); 8 waves, each owns 16 q rows.  Online softmax, causal.
__global__ __launch_bounds__(256)
void flash_attn(const __bf16* __restrict__ qh, const __bf16* __restrict__ kh,
                const __bf16* __restrict__ vt, __bf16* __restrict__ attn) {
  __shared__ __bf16 Pl[8 * 16 * 32];       // per-wave P transpose staging
  const int h = blockIdx.x, kvh = h >> 2;  // GQA 4:1
  const int wave = threadIdx.x >> 5, lane = threadIdx.x & 31;
  const int qbase = blockIdx.y * 128 + wave * 16;
  const __bf16* Qb = qh + ((size_t)h * SEQ + qbase) * HD;
  const __bf16* Kb = kh + (size_t)kvh * SEQ * HD;
  const __bf16* Vb = vt + (size_t)kvh * HD * SEQ;
  __bf16* Pw = &Pl[wave * 16 * 32];
  const v16bf aq0 = load_frag16(Qb, HD);        // d 0..31
  const v16bf aq1 = load_frag16(Qb + 32, HD);   // d 32..63
  v8f o[4] = {{}, {}, {}, {}};
  float m[8], l[8];
  #pragma unroll
  for (int j = 0; j < 8; ++j) { m[j] = -1e30f; l[j] = 0.f; }
  const int c = lane & 15, hi8 = (lane >> 4) * 8;
  const float scale = 0.125f;                   // 1/sqrt(64)
  for (int kb = 0; kb < qbase + 16; kb += 32) {
    v8f s0 = {}, s1 = {};
    // QK^T: K rows serve directly as B fragments (layout matches)
    s0 = wmma_bf16(aq0, load_frag16(Kb + (size_t)kb * HD, HD), s0);
    s0 = wmma_bf16(aq1, load_frag16(Kb + (size_t)kb * HD + 32, HD), s0);
    s1 = wmma_bf16(aq0, load_frag16(Kb + (size_t)(kb + 16) * HD, HD), s1);
    s1 = wmma_bf16(aq1, load_frag16(Kb + (size_t)(kb + 16) * HD + 32, HD), s1);
    const bool dm = (kb + 31 > qbase);
    #pragma unroll
    for (int j = 0; j < 8; ++j) {
      float v0 = s0[j] * scale, v1 = s1[j] * scale;
      if (dm) {
        const int row = qbase + hi8 + j;
        if (kb + c > row) v0 = -1e30f;
        if (kb + 16 + c > row) v1 = -1e30f;
      }
      float mx = fmaxf(v0, v1);
      #pragma unroll
      for (int off = 8; off; off >>= 1) mx = fmaxf(mx, __shfl_xor(mx, off));
      const float mn = fmaxf(m[j], mx);
      const float al = __expf(m[j] - mn);
      const float e0 = __expf(v0 - mn), e1 = __expf(v1 - mn);
      float rs = e0 + e1;
      #pragma unroll
      for (int off = 8; off; off >>= 1) rs += __shfl_xor(rs, off);
      l[j] = l[j] * al + rs;
      m[j] = mn;
      #pragma unroll
      for (int t = 0; t < 4; ++t) o[t][j] *= al;
      Pw[(hi8 + j) * 32 + c]      = (__bf16)e0;   // C-layout -> row-major P
      Pw[(hi8 + j) * 32 + 16 + c] = (__bf16)e1;
    }
    const v16bf pa = load_frag16(Pw, 32);         // A fragment of P
    #pragma unroll
    for (int t = 0; t < 4; ++t) {
      const v16bf bv = load_frag16(Vb + (size_t)(t * 16) * SEQ + kb, SEQ);
      o[t] = wmma_bf16(pa, bv, o[t]);
    }
  }
  #pragma unroll
  for (int t = 0; t < 4; ++t) {
    #pragma unroll
    for (int j = 0; j < 8; ++j) {
      const float inv = 1.f / l[j];
      attn[(size_t)(qbase + hi8 + j) * (NH * HD) + h * HD + t * 16 + c] =
          (__bf16)(o[t][j] * inv);
    }
  }
}

// ---------------------------------------------------------------- misc
__global__ __launch_bounds__(256)
void silu_mul(const float* __restrict__ gu, __bf16* __restrict__ a, int n) {
  const int i = blockIdx.x * 256 + threadIdx.x;
  if (i < n) {
    const int s = i >> 13, f = i & (FFD - 1);   // FFD = 8192
    const float gv = gu[(size_t)s * (2 * FFD) + f];
    const float uv = gu[(size_t)s * (2 * FFD) + FFD + f];
    a[i] = (__bf16)(gv / (1.f + __expf(-gv)) * uv);
  }
}

__global__ __launch_bounds__(256)
void copy_f32(const float* __restrict__ s, float* __restrict__ d, int n) {
  const int i = blockIdx.x * 256 + threadIdx.x;
  if (i < n) d[i] = s[i];
}

// ---------------------------------------------------------------- driver
extern "C" void kernel_launch(void* const* d_in, const int* in_sizes, int n_in,
                              void* d_out, int out_size, void* d_ws,
                              size_t ws_size, hipStream_t stream) {
  const int*   ids   = (const int*)d_in[0];
  const float* emb   = (const float*)d_in[1];
  const float* wq    = (const float*)d_in[2];
  const float* wk    = (const float*)d_in[3];
  const float* wv    = (const float*)d_in[4];
  const float* wo    = (const float*)d_in[5];
  const float* wgate = (const float*)d_in[6];
  const float* wup   = (const float*)d_in[7];
  const float* wdown = (const float*)d_in[8];
  const float* ln1   = (const float*)d_in[9];
  const float* ln2   = (const float*)d_in[10];
  float* out = (float*)d_out;

  char* p = (char*)d_ws;
  auto alloc = [&](size_t bytes) {
    void* r = (void*)p;
    p += (bytes + 255) & ~(size_t)255;
    return r;
  };
  // activations
  float*  x    = (float*)alloc((size_t)SEQ * DMODEL * 4);
  __bf16* hbuf = (__bf16*)alloc((size_t)SEQ * DMODEL * 2);
  float*  qkvf = (float*)alloc((size_t)SEQ * 3072 * 4);
  __bf16* qhb  = (__bf16*)alloc((size_t)NH * SEQ * HD * 2);
  __bf16* khb  = (__bf16*)alloc((size_t)NKV * SEQ * HD * 2);
  __bf16* vtb  = (__bf16*)alloc((size_t)NKV * HD * SEQ * 2);
  __bf16* atb  = (__bf16*)alloc((size_t)SEQ * NH * HD * 2);
  float*  gu   = (float*)alloc((size_t)SEQ * 2 * FFD * 4);
  __bf16* abuf = (__bf16*)alloc((size_t)SEQ * FFD * 2);
  // transposed bf16 weights (reused across layers)
  __bf16* wqkvT = (__bf16*)alloc((size_t)3072 * DMODEL * 2);      // [3072][2048]
  __bf16* woT   = (__bf16*)alloc((size_t)DMODEL * DMODEL * 2);    // [2048][2048]
  __bf16* wguT  = (__bf16*)alloc((size_t)2 * FFD * DMODEL * 2);   // [16384][2048]
  __bf16* wdT   = (__bf16*)alloc((size_t)DMODEL * FFD * 2);       // [2048][8192]

  embed_gather<<<SEQ, 256, 0, stream>>>(ids, emb, x);

  for (int l = 0; l < LAYERS; ++l) {
    const float* Wq = wq + (size_t)l * DMODEL * NH * HD;
    const float* Wk = wk + (size_t)l * DMODEL * NKV * HD;
    const float* Wv = wv + (size_t)l * DMODEL * NKV * HD;
    const float* Wo = wo + (size_t)l * NH * HD * DMODEL;
    const float* Wg = wgate + (size_t)l * DMODEL * FFD;
    const float* Wu = wup + (size_t)l * DMODEL * FFD;
    const float* Wd = wdown + (size_t)l * FFD * DMODEL;

    // weight transpose + fp32->bf16 (streamed once; packs qkv and gate|up)
    transpose_bf16<<<dim3(2048 / 32, DMODEL / 32), 256, 0, stream>>>(
        Wq, wqkvT, DMODEL, 2048);
    transpose_bf16<<<dim3(512 / 32, DMODEL / 32), 256, 0, stream>>>(
        Wk, wqkvT + (size_t)2048 * DMODEL, DMODEL, 512);
    transpose_bf16<<<dim3(512 / 32, DMODEL / 32), 256, 0, stream>>>(
        Wv, wqkvT + (size_t)2560 * DMODEL, DMODEL, 512);
    transpose_bf16<<<dim3(DMODEL / 32, 2048 / 32), 256, 0, stream>>>(
        Wo, woT, 2048, DMODEL);
    transpose_bf16<<<dim3(FFD / 32, DMODEL / 32), 256, 0, stream>>>(
        Wg, wguT, DMODEL, FFD);
    transpose_bf16<<<dim3(FFD / 32, DMODEL / 32), 256, 0, stream>>>(
        Wu, wguT + (size_t)FFD * DMODEL, DMODEL, FFD);
    transpose_bf16<<<dim3(DMODEL / 32, FFD / 32), 256, 0, stream>>>(
        Wd, wdT, FFD, DMODEL);

    // attention block
    rmsnorm_bf16<<<SEQ, 256, 0, stream>>>(x, ln1 + (size_t)l * DMODEL, hbuf);
    gemm_tn<false><<<dim3(3072 / 128, SEQ / 128), 256, 0, stream>>>(
        hbuf, wqkvT, qkvf, nullptr, SEQ, 3072, DMODEL);
    rope_pack<<<dim3(SEQ, NH + 2 * NKV), 64, 0, stream>>>(qkvf, qhb, khb, vtb);
    flash_attn<<<dim3(NH, SEQ / 128), 256, 0, stream>>>(qhb, khb, vtb, atb);
    gemm_tn<true><<<dim3(DMODEL / 128, SEQ / 128), 256, 0, stream>>>(
        atb, woT, x, x, SEQ, DMODEL, NH * HD);

    // MLP block
    rmsnorm_bf16<<<SEQ, 256, 0, stream>>>(x, ln2 + (size_t)l * DMODEL, hbuf);
    gemm_tn<false><<<dim3((2 * FFD) / 128, SEQ / 128), 256, 0, stream>>>(
        hbuf, wguT, gu, nullptr, SEQ, 2 * FFD, DMODEL);
    const int nffd = SEQ * FFD;
    silu_mul<<<(nffd + 255) / 256, 256, 0, stream>>>(gu, abuf, nffd);
    gemm_tn<true><<<dim3(DMODEL / 128, SEQ / 128), 256, 0, stream>>>(
        abuf, wdT, x, x, SEQ, DMODEL, FFD);
  }

  const int nout = SEQ * DMODEL;
  copy_f32<<<(nout + 255) / 256, 256, 0, stream>>>(x, out, nout);
}
